// GloAttnConv_27719718928663
// MI455X (gfx1250) — matrix operand order repaired and checked by reading, not compiled, not run on hardware
//
#include <hip/hip_runtime.h>

// GloAttnConv for MI455X (gfx1250, wave32, WMMA).
// H=2 heads, C=128 channels, K_ORDER=4 hops, BETA=0.5.

typedef __attribute__((ext_vector_type(16))) _Float16 v16h;
typedef __attribute__((ext_vector_type(8)))  _Float16 v8h;
typedef __attribute__((ext_vector_type(8)))  float    v8f;

union V16 { v16h v; v8h h[2]; };

static __device__ inline v8f wmma32(v16h a, v16h b, v8f c) {
  // D = A(16x32 f16) * B(32x16 f16) + C(16x16 f32)
  return __builtin_amdgcn_wmma_f32_16x16x32_f16(false, a, false, b, (short)0, c,
                                                false, false);
}

// A fragment from f16 row-major source; base already points at (row0, k0).
// CDNA5 A layout: lane L -> row L%16; lanes 0-15 hold K {0..7,16..23},
// lanes 16-31 hold K {8..15,24..31}.
static __device__ inline v16h loadA_f16(const _Float16* base, long ld, int lane) {
  int l = lane & 15, hi = lane >> 4;
  const _Float16* p = base + (long)l * ld + hi * 8;
  V16 u;
  u.h[0] = *(const v8h*)(p);
  u.h[1] = *(const v8h*)(p + 16);
  return u.v;
}

// A fragment from f32 row-major source with per-row validity guard.
// p already points at (this lane's row, kb).
static __device__ inline v16h loadA_f32(const float* p, bool ok) {
  v16h a;
#pragma unroll
  for (int e = 0; e < 8; ++e) {
    a[e]     = ok ? (_Float16)p[e]      : (_Float16)0.f;
    a[e + 8] = ok ? (_Float16)p[16 + e] : (_Float16)0.f;
  }
  return a;
}

// B fragment from B^T-style storage bt[n][k] (leading dim ld halves):
// lane L -> column N = L%16; lanes 0-15 hold K {0..7,16..23}, lanes 16-31 hold K {8..15,24..31}.
static __device__ inline v16h loadB_t(const _Float16* bt, int ld, int lane) {
  int n = lane & 15, hi = lane >> 4;
  const _Float16* p = bt + (long)n * ld + hi * 8;
  V16 u;
  u.h[0] = *(const v8h*)(p);
  u.h[1] = *(const v8h*)(p + 16);
  return u.v;
}

// ---------------------------------------------------------------- small kernels

__global__ void k_deg(const int* __restrict__ ei, long E, float* __restrict__ deg) {
  long i = (long)blockIdx.x * 256 + threadIdx.x;
  if (i < E) atomicAdd(&deg[ei[E + i]], 1.0f);
}

__global__ void k_deginv(float* __restrict__ deg, int N) {
  long i = (long)blockIdx.x * 256 + threadIdx.x;
  if (i < N) { float d = deg[i]; deg[i] = d > 0.f ? 1.f / d : 0.f; }
}

// prev[n][h][c] = x[n][c] for both heads; zero for pad rows.
__global__ void k_prev_init(const float* __restrict__ x, float* __restrict__ prev,
                            int N, int Npad) {
  long i = (long)blockIdx.x * 256 + threadIdx.x;
  long total = (long)Npad * 128;
  if (i < total) {
    long n = i >> 7; int c = (int)(i & 127);
    float v = (n < N) ? x[i] : 0.f;
    prev[n * 256 + c] = v;
    prev[n * 256 + 128 + c] = v;
  }
}

__global__ void k_out_init(const float* __restrict__ b, float* __restrict__ out,
                           long total) {
  long i = (long)blockIdx.x * 256 + threadIdx.x;
  if (i < total) out[i] = b[i & 127];
}

__global__ void k_scale(float* __restrict__ out, long total) {
  long i = (long)blockIdx.x * 256 + threadIdx.x;
  if (i < total) out[i] *= 0.5f;  // /H with H=2
}

// GCN scatter: one block per edge, 256 channels.
__global__ __launch_bounds__(256) void k_gcn(const int* __restrict__ ei, long E,
                                             const float* __restrict__ prev,
                                             float* __restrict__ cur) {
  long e = blockIdx.x;
  int r = ei[e];
  int c = ei[E + e];
  int t = threadIdx.x;
  atomicAdd(&cur[(long)c * 256 + t], prev[(long)r * 256 + t]);
}

// ------------------------------------------------------------------- QK kernel
// grid: (Npad/128, H, {q,k}); 8 waves, each owns a 16-node block, full 128-col head.
__global__ __launch_bounds__(256) void k_qk(const float* __restrict__ x,
    const float* __restrict__ Wq, const float* __restrict__ bq,
    const float* __restrict__ Wk, const float* __restrict__ bk,
    _Float16* __restrict__ qs, _Float16* __restrict__ ksT,
    float* __restrict__ ks_sum, int N, int Npad) {
  const int LDW = 136;                    // padded: 2-way-max LDS bank conflicts
  __shared__ _Float16 ldsW[128 * 136];    // [col][m] transposed weight tile
  int h = blockIdx.y;
  int isK = blockIdx.z;
  const float* W = isK ? Wk : Wq;
  const float* bias = isK ? bk : bq;

  for (int idx = threadIdx.x; idx < 128 * 128; idx += 256) {
    int col = idx & 127, m = idx >> 7;    // col fast -> coalesced global read
    ldsW[col * LDW + m] = (_Float16)W[m * 256 + h * 128 + col];
  }
  __syncthreads();

  int lane = threadIdx.x & 31, w = threadIdx.x >> 5;
  int l = lane & 15, hi = lane >> 4;
  long r0 = (long)blockIdx.x * 128 + w * 16;
  long rowA = r0 + l;

  v16h A[4];
#pragma unroll
  for (int f = 0; f < 4; ++f)
    A[f] = loadA_f32(x + rowA * 128 + f * 32 + hi * 8, rowA < (long)N);

  v8f acc[8];
#pragma unroll
  for (int ct = 0; ct < 8; ++ct) {
    v8f c;
#pragma unroll
    for (int v = 0; v < 8; ++v) c[v] = 0.f;
#pragma unroll
    for (int f = 0; f < 4; ++f)
      c = wmma32(A[f], loadB_t(ldsW + ct * 16 * LDW + f * 32, LDW, lane), c);
    float bv = bias[h * 128 + ct * 16 + l];
#pragma unroll
    for (int v = 0; v < 8; ++v) c[v] += bv;
    acc[ct] = c;
  }

  // per-row L2 norm: reduce across the 16 lanes of each half-wave
  float rn[8];
#pragma unroll
  for (int v = 0; v < 8; ++v) {
    float s = 0.f;
#pragma unroll
    for (int ct = 0; ct < 8; ++ct) s += acc[ct][v] * acc[ct][v];
    s += __shfl_xor(s, 1);
    s += __shfl_xor(s, 2);
    s += __shfl_xor(s, 4);
    s += __shfl_xor(s, 8);
    long rowv = r0 + hi * 8 + v;
    rn[v] = (rowv < (long)N && s > 0.f) ? rsqrtf(s) : 0.f;  // pad rows -> 0
  }
#pragma unroll
  for (int ct = 0; ct < 8; ++ct)
#pragma unroll
    for (int v = 0; v < 8; ++v) acc[ct][v] *= rn[v];

  if (!isK) {
    // qs row-major [Npad][256] f16
#pragma unroll
    for (int ct = 0; ct < 8; ++ct) {
      int col = h * 128 + ct * 16 + l;
#pragma unroll
      for (int v = 0; v < 8; ++v) {
        long rowv = r0 + hi * 8 + v;
        qs[rowv * 256 + col] = (_Float16)acc[ct][v];
      }
    }
  } else {
    // ksT [H][128][Npad] f16 (node dim contiguous -> 16B packed stores)
#pragma unroll
    for (int ct = 0; ct < 8; ++ct) {
      int m = ct * 16 + l;
      v8h pk;
      float s = 0.f;
#pragma unroll
      for (int v = 0; v < 8; ++v) { pk[v] = (_Float16)acc[ct][v]; s += acc[ct][v]; }
      *(v8h*)(ksT + ((long)h * 128 + m) * Npad + r0 + hi * 8) = pk;
      atomicAdd(&ks_sum[h * 128 + m], s);
    }
  }
}

// ------------------------------------------------------------------ kvs kernel
// kvs[h] += ksT[h] (128 x chunk) * prev (chunk x 128); also vs_sum partials.
__global__ __launch_bounds__(256) void k_kvs(const _Float16* __restrict__ ksT,
    const float* __restrict__ prev, float* __restrict__ kvs,
    float* __restrict__ vs_sum, int Npad, int chunk) {
  const int LDB = 40;                     // padded: conflict-free strided reads
  __shared__ _Float16 ldsB[128 * 40];     // [d][node] transposed prev tile
  int h = blockIdx.y;
  long c0 = (long)blockIdx.x * chunk;
  long c1 = c0 + chunk; if (c1 > (long)Npad) c1 = Npad;
  int lane = threadIdx.x & 31, w = threadIdx.x >> 5;
  int l = lane & 15, hi = lane >> 4;

  v8f acc[8];
#pragma unroll
  for (int dt = 0; dt < 8; ++dt)
#pragma unroll
    for (int v = 0; v < 8; ++v) acc[dt][v] = 0.f;
  float vpart = 0.f;

  for (long n0 = c0; n0 < c1; n0 += 32) {
    __syncthreads();
    for (int idx = threadIdx.x; idx < 4096; idx += 256) {
      int d = idx & 127, n = idx >> 7;    // d fast -> coalesced global read
      float v = prev[(n0 + n) * 256 + h * 128 + d];
      ldsB[d * LDB + n] = (_Float16)v;
      vpart += v;                          // this thread's d is fixed = tid&127
    }
    __syncthreads();
    v16h A = loadA_f16(ksT + ((long)h * 128 + w * 16) * Npad + n0, Npad, lane);
#pragma unroll
    for (int dt = 0; dt < 8; ++dt)
      acc[dt] = wmma32(A, loadB_t(ldsB + dt * 16 * LDB, LDB, lane), acc[dt]);
  }

#pragma unroll
  for (int dt = 0; dt < 8; ++dt)
#pragma unroll
    for (int v = 0; v < 8; ++v)
      atomicAdd(&kvs[h * 16384 + (w * 16 + hi * 8 + v) * 128 + dt * 16 + l],
                acc[dt][v]);
  atomicAdd(&vs_sum[h * 128 + (threadIdx.x & 127)], vpart);
}

// -------------------------------------------------------------- combine kernel
// cur = BETA*deg_inv*agg + (qs@kvs + vs_sum) / (qs.ks_sum + N), in place on cur.
__global__ __launch_bounds__(256) void k_combine(const _Float16* __restrict__ qs,
    const float* __restrict__ kvs, const float* __restrict__ vs_sum,
    const float* __restrict__ ks_sum, const float* __restrict__ deg_inv,
    float* __restrict__ cur, int N, int Npad, float fN) {
  const int LDK = 136;
  __shared__ _Float16 ldsKvs[128 * 136];  // [d][m] transposed kvs tile
  __shared__ float ldsKs[128];
  int h = blockIdx.y;
  for (int idx = threadIdx.x; idx < 16384; idx += 256) {
    int d = idx & 127, m = idx >> 7;
    ldsKvs[d * LDK + m] = (_Float16)kvs[h * 16384 + m * 128 + d];
  }
  if (threadIdx.x < 128) ldsKs[threadIdx.x] = ks_sum[h * 128 + threadIdx.x];
  __syncthreads();

  int lane = threadIdx.x & 31, w = threadIdx.x >> 5;
  int l = lane & 15, hi = lane >> 4;
  long r0 = (long)blockIdx.x * 128 + w * 16;

  v16h A[4];
  float dpart = 0.f;
#pragma unroll
  for (int f = 0; f < 4; ++f) {
    A[f] = loadA_f16(qs + r0 * 256 + h * 128 + f * 32, 256, lane);
#pragma unroll
    for (int e = 0; e < 8; ++e) {
      dpart += (float)A[f][e]     * ldsKs[f * 32 + hi * 8 + e];
      dpart += (float)A[f][e + 8] * ldsKs[f * 32 + hi * 8 + 16 + e];
    }
  }
  float dden = dpart + __shfl_xor(dpart, 16);  // lane M (M<16) holds den(row M)
  float dm[8];
#pragma unroll
  for (int v = 0; v < 8; ++v) dm[v] = __shfl(dden, hi * 8 + v) + fN;

  v8f acc[8];
#pragma unroll
  for (int dt = 0; dt < 8; ++dt) {
#pragma unroll
    for (int v = 0; v < 8; ++v) acc[dt][v] = 0.f;
#pragma unroll
    for (int f = 0; f < 4; ++f)
      acc[dt] = wmma32(A[f], loadB_t(ldsKvs + dt * 16 * LDK + f * 32, LDK, lane),
                       acc[dt]);
  }

#pragma unroll
  for (int dt = 0; dt < 8; ++dt) {
    int d = dt * 16 + l;
    float vsv = vs_sum[h * 128 + d];
#pragma unroll
    for (int v = 0; v < 8; ++v) {
      long rowv = r0 + hi * 8 + v;
      if (rowv < (long)N) {
        long off = rowv * 256 + h * 128 + d;
        float agg = cur[off];
        cur[off] = 0.5f * deg_inv[rowv] * agg + (acc[dt][v] + vsv) / dm[v];
      }
      // pad rows: stay zero (memset before scatter, never written here)
    }
  }
}

// -------------------------------------------------------------- output projection
// out[n] += sum_h term[n,h,:] @ Wo[h*640 + k*128 .. +128, :]   (head-major concat!)
__global__ __launch_bounds__(256) void k_outproj(const float* __restrict__ term,
    const float* __restrict__ xsrc, int isX, const float* __restrict__ Wo,
    float* __restrict__ out, int kterm, int N, int Npad) {
  const int LDO = 136;
  __shared__ _Float16 ldsWo[128 * 136];   // [nout][c] transposed Wo tile, per head
  int lane = threadIdx.x & 31, w = threadIdx.x >> 5;
  int l = lane & 15, hi = lane >> 4;
  long r0 = (long)blockIdx.x * 128 + w * 16;

  v8f acc[8];
#pragma unroll
  for (int nt = 0; nt < 8; ++nt)
#pragma unroll
    for (int v = 0; v < 8; ++v) acc[nt][v] = 0.f;

  for (int hh = 0; hh < 2; ++hh) {
    __syncthreads();
    for (int idx = threadIdx.x; idx < 16384; idx += 256) {
      int nout = idx & 127, c = idx >> 7;  // nout fast -> coalesced global read
      ldsWo[nout * LDO + c] =
          (_Float16)Wo[((long)hh * 640 + kterm * 128 + c) * 128 + nout];
    }
    __syncthreads();
    long row = r0 + l;
#pragma unroll
    for (int f = 0; f < 4; ++f) {
      v16h A;
      if (isX) {
        A = loadA_f32(xsrc + row * 128 + f * 32 + hi * 8, row < (long)N);
      } else {
        A = loadA_f32(term + row * 256 + hh * 128 + f * 32 + hi * 8, true);
      }
#pragma unroll
      for (int nt = 0; nt < 8; ++nt)
        acc[nt] = wmma32(A, loadB_t(ldsWo + nt * 16 * LDO + f * 32, LDO, lane),
                         acc[nt]);
    }
  }

#pragma unroll
  for (int nt = 0; nt < 8; ++nt)
#pragma unroll
    for (int v = 0; v < 8; ++v) {
      long rowv = r0 + hi * 8 + v;
      if (rowv < (long)N) out[rowv * 128 + nt * 16 + l] += acc[nt][v];
    }
}

// ----------------------------------------------------------------------- launch

extern "C" void kernel_launch(void* const* d_in, const int* in_sizes, int n_in,
                              void* d_out, int out_size, void* d_ws, size_t ws_size,
                              hipStream_t stream) {
  (void)n_in; (void)out_size; (void)ws_size;
  const float* x    = (const float*)d_in[0];
  const int*   ei   = (const int*)d_in[1];     // [2][E] (row; col)
  const float* Wq_w = (const float*)d_in[2];
  const float* Wq_b = (const float*)d_in[3];
  const float* Wk_w = (const float*)d_in[4];
  const float* Wk_b = (const float*)d_in[5];
  const float* Wo_w = (const float*)d_in[6];
  const float* Wo_b = (const float*)d_in[7];
  float* out = (float*)d_out;

  int  N = in_sizes[0] / 128;
  long E = (long)in_sizes[1] / 2;
  int  Npad = (N + 127) & ~127;

  // workspace carve (256B aligned): ~308 MB total
  char* base = (char*)d_ws;
  size_t off = 0;
  auto carve = [&](size_t bytes) {
    void* p = base + off;
    off += (bytes + 255) & ~(size_t)255;
    return p;
  };
  float*     deg  = (float*)carve((size_t)Npad * 4);
  _Float16*  qs   = (_Float16*)carve((size_t)Npad * 256 * 2);
  _Float16*  ksT  = (_Float16*)carve((size_t)Npad * 256 * 2);
  float*     bufA = (float*)carve((size_t)Npad * 256 * 4);
  float*     bufB = (float*)carve((size_t)Npad * 256 * 4);
  float*     kvs  = (float*)carve(2 * 128 * 128 * 4);
  float*     vs   = (float*)carve(256 * 4);
  float*     kss  = (float*)carve(256 * 4);

  long outTotal = (long)N * 128;

  hipMemsetAsync(deg, 0, (size_t)Npad * 4, stream);
  hipMemsetAsync(kss, 0, 256 * 4, stream);

  k_deg<<<dim3((unsigned)((E + 255) / 256)), 256, 0, stream>>>(ei, E, deg);
  k_deginv<<<dim3((N + 255) / 256), 256, 0, stream>>>(deg, N);

  k_qk<<<dim3(Npad / 128, 2, 2), 256, 0, stream>>>(x, Wq_w, Wq_b, Wk_w, Wk_b,
                                                   qs, ksT, kss, N, Npad);

  k_prev_init<<<dim3((unsigned)(((size_t)Npad * 128 + 255) / 256)), 256, 0,
                stream>>>(x, bufA, N, Npad);

  k_out_init<<<dim3((unsigned)((outTotal + 255) / 256)), 256, 0, stream>>>(
      Wo_b, out, outTotal);
  k_outproj<<<dim3(Npad / 128), 256, 0, stream>>>(nullptr, x, 1, Wo_w, out, 0,
                                                  N, Npad);

  float* prev = bufA;
  float* cur  = bufB;
  const int CHUNK = 2048;
  for (int k = 1; k <= 4; ++k) {
    hipMemsetAsync(kvs, 0, 2 * 128 * 128 * 4, stream);
    hipMemsetAsync(vs, 0, 256 * 4, stream);
    k_kvs<<<dim3((Npad + CHUNK - 1) / CHUNK, 2), 256, 0, stream>>>(
        ksT, prev, kvs, vs, Npad, CHUNK);

    hipMemsetAsync(cur, 0, (size_t)Npad * 256 * 4, stream);
    k_gcn<<<dim3((unsigned)E), 256, 0, stream>>>(ei, E, prev, cur);

    k_combine<<<dim3(Npad / 128, 2), 256, 0, stream>>>(qs, kvs, vs, kss, deg,
                                                       cur, N, Npad, (float)N);
    k_outproj<<<dim3(Npad / 128), 256, 0, stream>>>(cur, nullptr, 0, Wo_w, out,
                                                    k, N, Npad);
    float* t = prev; prev = cur; cur = t;
  }

  k_scale<<<dim3((unsigned)((outTotal + 255) / 256)), 256, 0, stream>>>(out,
                                                                        outTotal);
}
